// FullyConnectedTensorProductConv_80195629350947
// MI455X (gfx1250) — matrix operand order repaired and checked by reading, not compiled, not run on hardware
//
#include <hip/hip_runtime.h>
#include <math.h>

typedef __attribute__((ext_vector_type(16))) __bf16 v16bf;
typedef __attribute__((ext_vector_type(8)))  __bf16 v8bf;
typedef __attribute__((ext_vector_type(8)))  float  v8f;

#define MULT      16
#define N_NODES   10000
#define N_EDGES   160000
#define TILE_E    16

__device__ __forceinline__ float gelu_exact(float x) {
  return 0.5f * x * (1.0f + erff(x * 0.70710678118654752f));
}

// ---------------------------------------------------------------------------
// Prep: transpose + convert W1 (128x128) and W2 (128x1024) to bf16 col-major
// so WMMA B-fragments become two contiguous 16-byte per-lane loads.
// ---------------------------------------------------------------------------
__global__ void prep_weights(const float* __restrict__ W1,
                             const float* __restrict__ W2,
                             __bf16* __restrict__ W1T,
                             __bf16* __restrict__ W2T) {
  int i = blockIdx.x * blockDim.x + threadIdx.x;
  if (i < 128 * 128) {
    int n = i >> 7, k = i & 127;
    W1T[n * 128 + k] = (__bf16)W1[k * 128 + n];
  }
  int j = i - 128 * 128;
  if (j >= 0 && j < 1024 * 128) {
    int n = j >> 7, k = j & 127;
    W2T[n * 128 + k] = (__bf16)W2[k * 1024 + n];
  }
}

__global__ void zero_buffers(float* __restrict__ out, float* __restrict__ cnt) {
  int i = blockIdx.x * blockDim.x + threadIdx.x;
  if (i < N_NODES * 64) out[i] = 0.0f;
  if (i < N_NODES)      cnt[i] = 0.0f;
}

// ---------------------------------------------------------------------------
// Fused per-edge-tile kernel: MLP1 (WMMA) -> MLP2 (WMMA, 16x1024 tile kept in
// LDS, never touches HBM) -> equivariant tensor product -> atomic scatter.
// One 256-thread block (8 wave32) handles 16 edges.
// ---------------------------------------------------------------------------
__launch_bounds__(256)
__global__ void fused_edge_kernel(
    const float* __restrict__ src_features,   // (N_NODES, 64)
    const float* __restrict__ edge_sh,        // (E, 4)
    const float* __restrict__ edge_emb,       // (E, 128)
    const __bf16* __restrict__ W1T,           // (128, 128) col-major bf16
    const float* __restrict__ b1,             // (128,)
    const __bf16* __restrict__ W2T,           // (1024, 128) col-major bf16
    const float* __restrict__ b2,             // (1024,)
    const int* __restrict__ src,
    const int* __restrict__ dst,
    float* __restrict__ out,                  // (N_NODES, 64) accumulated
    float* __restrict__ cnt)                  // (N_NODES,)
{
  __shared__ __attribute__((aligned(32))) __bf16 sX[TILE_E * 128];   // emb bf16
  __shared__ __attribute__((aligned(32))) __bf16 sH[TILE_E * 128];   // h bf16
  __shared__ __attribute__((aligned(16))) float  sTW[TILE_E * 1024]; // tp weights
  __shared__ float sXF[TILE_E * 64];                                 // gathered x
  __shared__ float sSH[TILE_E * 4];                                  // sh

  const int tid  = threadIdx.x;
  const int wave = tid >> 5;
  const int lane = tid & 31;
  const int e0   = blockIdx.x * TILE_E;

  // Stage edge_emb tile (f32 -> bf16)
  for (int i = tid; i < TILE_E * 128; i += 256)
    sX[i] = (__bf16)edge_emb[(size_t)e0 * 128 + i];
  // Stage gathered src features
  for (int i = tid; i < TILE_E * 64; i += 256) {
    int e = i >> 6, c = i & 63;
    sXF[i] = src_features[(size_t)src[e0 + e] * 64 + c];
  }
  if (tid < TILE_E * 4) sSH[tid] = edge_sh[(size_t)e0 * 4 + tid];
  __syncthreads();

  // WMMA lane decomposition (ISA 7.12.2 layouts, wave32):
  //   A 16x32 bf16: lane L & L+16 hold row (L&15); khalf=lane>>4 selects K subset
  //   B 32x16 bf16: col = lane&15; element e<8 -> K = khalf*8 + e, e>=8 -> K = 16 + khalf*8 + (e-8)
  //   C f32 16x16 : VGPR r -> M = khalf*8 + r, N = lane&15
  const int col   = lane & 15;
  const int khalf = lane >> 4;
  const int mbase = khalf * 8;

  // ---- GEMM1: h = gelu(emb @ W1 + b1); each wave -> one 16x16 tile --------
  {
    const int n0 = wave * 16;
    v8f acc = {};
#pragma unroll
    for (int kk = 0; kk < 4; ++kk) {
      union { v16bf v; v8bf h[2]; } A;
      const int ka = kk * 32 + khalf * 8;
      A.h[0] = *reinterpret_cast<const v8bf*>(&sX[col * 128 + ka]);
      A.h[1] = *reinterpret_cast<const v8bf*>(&sX[col * 128 + ka + 16]);
      union { v16bf v; v8bf h[2]; } Bu;
      Bu.h[0] = *reinterpret_cast<const v8bf*>(W1T + (size_t)(n0 + col) * 128 + ka);
      Bu.h[1] = *reinterpret_cast<const v8bf*>(W1T + (size_t)(n0 + col) * 128 + ka + 16);
      v16bf B = Bu.v;
      acc = __builtin_amdgcn_wmma_f32_16x16x32_bf16(
          false, A.v, false, B, (short)0, acc, false, false);
    }
    const float bias = b1[n0 + col];
#pragma unroll
    for (int r = 0; r < 8; ++r) {
      float hv = gelu_exact(acc[r] + bias);
      sH[(mbase + r) * 128 + n0 + col] = (__bf16)hv;
    }
  }
  __syncthreads();

  // ---- GEMM2: tpw = h @ W2 + b2; each wave -> 8 of 64 tiles; f32 to LDS ---
  // A fragments depend only on kk: hoist all 4 into registers once and reuse
  // across the 8 output tiles (8x fewer LDS reads, shorter WMMA dep chains).
  {
    union { v16bf v; v8bf h[2]; } Afrag[4];
#pragma unroll
    for (int kk = 0; kk < 4; ++kk) {
      const int ka = kk * 32 + khalf * 8;
      Afrag[kk].h[0] = *reinterpret_cast<const v8bf*>(&sH[col * 128 + ka]);
      Afrag[kk].h[1] = *reinterpret_cast<const v8bf*>(&sH[col * 128 + ka + 16]);
    }
#pragma unroll
    for (int t = 0; t < 8; ++t) {
      const int n0 = wave * 16 + t * 128;
      v8f acc = {};
#pragma unroll
      for (int kk = 0; kk < 4; ++kk) {
        union { v16bf v; v8bf h[2]; } Bu;
        Bu.h[0] = *reinterpret_cast<const v8bf*>(W2T + (size_t)(n0 + col) * 128 + kk * 32 + khalf * 8);
        Bu.h[1] = *reinterpret_cast<const v8bf*>(W2T + (size_t)(n0 + col) * 128 + kk * 32 + khalf * 8 + 16);
        v16bf B = Bu.v;
        acc = __builtin_amdgcn_wmma_f32_16x16x32_bf16(
            false, Afrag[kk].v, false, B, (short)0, acc, false, false);
      }
      const float bias = b2[n0 + col];
#pragma unroll
      for (int r = 0; r < 8; ++r)
        sTW[(mbase + r) * 1024 + n0 + col] = acc[r] + bias;
    }
  }
  __syncthreads();

  // ---- Tensor product + scatter: 16 threads per edge, one output w each ---
  {
    const float INV_SQRT3 = 0.57735026918962576f;
    const float ALPHA     = 0.17677669529663689f;  // 1/sqrt(2*MUL)
    const int e = tid >> 4;
    const int w = tid & 15;
    const float* xf = &sXF[e * 64];
    const float* tw = &sTW[e * 1024];
    const float sh0 = sSH[e * 4 + 0];
    const float s1x = sSH[e * 4 + 1];
    const float s1y = sSH[e * 4 + 2];
    const float s1z = sSH[e * 4 + 3];
    float o0 = 0.f, oA = 0.f, oB = 0.f, oC = 0.f;
#pragma unroll
    for (int u = 0; u < MULT; ++u) {
      const float su = xf[u];
      const float v0 = xf[16 + u * 3 + 0];
      const float v1 = xf[16 + u * 3 + 1];
      const float v2 = xf[16 + u * 3 + 2];
      const float w00  = tw[        u * 16 + w];
      const float w110 = tw[256  +  u * 16 + w];
      const float w01  = tw[512  +  u * 16 + w];
      const float w10  = tw[768  +  u * 16 + w];
      o0 += (su * sh0) * w00 +
            INV_SQRT3 * (v0 * s1x + v1 * s1y + v2 * s1z) * w110;
      const float t1 = su * w01;
      const float g  = sh0 * w10;
      oA += s1x * t1 + g * v0;
      oB += s1y * t1 + g * v1;
      oC += s1z * t1 + g * v2;
    }
    const int d = dst[e0 + e];
    float* op = out + (size_t)d * 64;
    atomicAdd(op + w,              ALPHA * o0);
    atomicAdd(op + 16 + w * 3 + 0, ALPHA * oA);
    atomicAdd(op + 16 + w * 3 + 1, ALPHA * oB);
    atomicAdd(op + 16 + w * 3 + 2, ALPHA * oC);
    if (w == 0) atomicAdd(cnt + d, 1.0f);
  }
}

__global__ void finalize_mean(float* __restrict__ out, const float* __restrict__ cnt) {
  int i = blockIdx.x * blockDim.x + threadIdx.x;
  if (i < N_NODES * 64)
    out[i] = out[i] / fmaxf(cnt[i >> 6], 1.0f);
}

// ---------------------------------------------------------------------------
extern "C" void kernel_launch(void* const* d_in, const int* in_sizes, int n_in,
                              void* d_out, int out_size, void* d_ws, size_t ws_size,
                              hipStream_t stream) {
  const float* src_features = (const float*)d_in[0];  // (10000, 64)
  const float* edge_sh      = (const float*)d_in[1];  // (160000, 4)
  const float* edge_emb     = (const float*)d_in[2];  // (160000, 128)
  const float* W1           = (const float*)d_in[3];  // (128, 128)
  const float* b1           = (const float*)d_in[4];  // (128,)
  const float* W2           = (const float*)d_in[5];  // (128, 1024)
  const float* b2           = (const float*)d_in[6];  // (1024,)
  const int*   src          = (const int*)d_in[7];
  const int*   dst          = (const int*)d_in[8];
  float* out = (float*)d_out;

  char* ws = (char*)d_ws;
  __bf16* W1T = (__bf16*)ws;                        // 32768 B
  __bf16* W2T = (__bf16*)(ws + 32768);              // 262144 B
  float*  cnt = (float*)(ws + 32768 + 262144);      // 40000 B

  prep_weights<<<(128 * 128 + 1024 * 128 + 255) / 256, 256, 0, stream>>>(W1, W2, W1T, W2T);
  zero_buffers<<<(N_NODES * 64 + 255) / 256, 256, 0, stream>>>(out, cnt);
  fused_edge_kernel<<<N_EDGES / TILE_E, 256, 0, stream>>>(
      src_features, edge_sh, edge_emb, W1T, b1, W2T, b2, src, dst, out, cnt);
  finalize_mean<<<(N_NODES * 64 + 255) / 256, 256, 0, stream>>>(out, cnt);
}
